// RaggedLogisticNormalSurrogate_35751307772544
// MI455X (gfx1250) — matrix operand order repaired and checked by reading, not compiled, not run on hardware
//
#include <hip/hip_runtime.h>
#include <math.h>

// Problem constants (match reference)
#define BN     256
#define INF_   256
#define DB     29
#define HID    64
#define NF     6
#define DMAX   8
#define HMAX   32
#define WMAX   32
#define KTOT   8192                    // DMAX*HMAX*WMAX
#define MC     16
#define BKTOT  ((size_t)BN * KTOT)     // 2097152
#define PE_DIM 36                      // 6*NF

// LDS strides (in halves / floats)
#define WST  80    // weight rows: 160B stride, 16/32B aligned sub-chunks
#define HSST 40    // h_samples rows: 80B stride
#define WFST 40    // Wflat staging rows
#define LST  20    // logits staging rows (floats)

typedef __attribute__((ext_vector_type(16))) _Float16 v16h;
typedef __attribute__((ext_vector_type(8)))  _Float16 v8h;
typedef __attribute__((ext_vector_type(8)))  float    v8f;

__device__ __forceinline__ v8f wmma16(v16h a, v16h b, v8f c) {
  // D = A(16x32 f16) * B(32x16 f16) + C(16x16 f32)
  return __builtin_amdgcn_wmma_f32_16x16x32_f16(false, a, false, b, (short)0, c,
                                                false, false);
}

// Load 16 contiguous halves (B fragment for one lane): two 16B LDS loads.
__device__ __forceinline__ v16h ld_contig16(const _Float16* p) {
  v8h lo = *(const v8h*)(p);
  v8h hi = *(const v8h*)(p + 8);
  v16h r;
#pragma unroll
  for (int i = 0; i < 8; ++i) { r[i] = lo[i]; r[i + 8] = hi[i]; }
  return r;
}

// A-fragment from a row stored [k]-contiguous in LDS.
// Layout: lane(half=0) holds k in {kb..kb+15}, half=1 holds kb+16..kb+31.
__device__ __forceinline__ v16h ld_afrag(const _Float16* rowp, int half, int kbase) {
  const _Float16* base = rowp + kbase + half * 16;
  v8h lo = *(const v8h*)(base);
  v8h hi = *(const v8h*)(base + 8);
  v16h r;
#pragma unroll
  for (int i = 0; i < 8; ++i) { r[i] = lo[i]; r[i + 8] = hi[i]; }
  return r;
}

__device__ __forceinline__ float gelu_f(float x) {
  return 0.5f * x * (1.0f + erff(x * 0.7071067811865476f));
}

__device__ __forceinline__ float axis_coord(int i, int s) {
  return (s > 1) ? (-1.0f + 2.0f * (float)i / (float)(s - 1)) : 0.0f;
}

// Fourier feature j (j<36): axis = j/12, t = j%12; t<6 -> sin(c*pi*2^t) else cos.
__device__ __forceinline__ float pe_val(float c0, float c1, float c2, int k) {
  if (k >= PE_DIM) return 0.0f;
  int axis = k / 12;
  int tt = k - axis * 12;
  int q = (tt < 6) ? tt : (tt - 6);
  float c = (axis == 0) ? c0 : ((axis == 1) ? c1 : c2);
  float ph = c * (3.14159265358979323846f * (float)(1 << q));
  return (tt < 6) ? __sinf(ph) : __cosf(ph);
}

// ---------------------------------------------------------------------------
// Kernel A: mu_h = feat @ mu_W.T + mu_b ; std_h = softplus(feat @ std_W.T + std_b)
// One 64-thread block per (b,d) slot.
// ---------------------------------------------------------------------------
__global__ __launch_bounds__(64) void k_mustd(
    const float* __restrict__ feat, const float* __restrict__ muW,
    const float* __restrict__ mub, const float* __restrict__ stdW,
    const float* __restrict__ stdb, float* __restrict__ mu_out,
    float* __restrict__ std_out) {
  int blk = blockIdx.x;
  int d = blk & 31, b = blk >> 5;
  if (d >= DB) return;
  int t = threadIdx.x;
  __shared__ float smu[64], sst[64];
  const float* f = feat + (size_t)b * INF_;
  const float* wm = muW + (size_t)d * INF_;
  const float* ws = stdW + (size_t)d * INF_;
  float am = 0.0f, as = 0.0f;
#pragma unroll
  for (int j = 0; j < 4; ++j) {
    int i = t + j * 64;
    float fv = f[i];
    am = fmaf(fv, wm[i], am);
    as = fmaf(fv, ws[i], as);
  }
  smu[t] = am; sst[t] = as;
  __syncthreads();
  for (int off = 32; off > 0; off >>= 1) {
    if (t < off) { smu[t] += smu[t + off]; sst[t] += sst[t + off]; }
    __syncthreads();
  }
  if (t == 0) {
    float m = smu[0] + mub[d];
    float x = sst[0] + stdb[d];
    float sp = (x > 0.0f) ? (x + log1pf(expf(-x))) : log1pf(expf(x));
    mu_out[b * DB + d] = m;
    std_out[b * DB + d] = sp;
  }
}

// ---------------------------------------------------------------------------
// Kernel B: fused Fourier-encode -> 3-layer MLP (WMMA f16) -> logits (WMMA)
// Block: 256 threads = 8 wave32; each wave owns a 16-point M-tile.
// Grid: BN * (KTOT/128) workgroups.
// Writes raw (masked) logits into d_out: zp rows [0..16*B*K) then mean row.
// ---------------------------------------------------------------------------
__global__ __launch_bounds__(256) void k_mlp_logits(
    const float* __restrict__ w1, const float* __restrict__ b1,
    const float* __restrict__ w2, const float* __restrict__ b2,
    const float* __restrict__ w3, const float* __restrict__ b3,
    const float* __restrict__ eps, const int* __restrict__ shapes,
    const float* __restrict__ logb, const float* __restrict__ mu,
    const float* __restrict__ stdv, float* __restrict__ out) {
  // Weight tiles in WMMA-B-friendly layout (row = output neuron n, cols = k).
  alignas(32) __shared__ _Float16 w1s[64 * WST];
  alignas(32) __shared__ _Float16 w2s[64 * WST];
  alignas(32) __shared__ _Float16 w3s[32 * WST];
  alignas(32) __shared__ _Float16 hss[32 * HSST];   // h_samples (s=0..15, 16=mu)
  alignas(32) __shared__ _Float16 hstage[8][16 * WST];  // per-wave h1/h2 staging (reused)
  alignas(32) __shared__ _Float16 wfs[8][16 * WFST];    // per-wave Wflat f16
  __shared__ float b1s[64], b2s[64], b3s[32];

  int t = threadIdx.x;
  int b = blockIdx.x >> 6;        // batch
  int tile = blockIdx.x & 63;     // 128-point tile within K

  // ---- cooperative staging (f32 -> f16) ----
  for (int idx = t; idx < 64 * 64; idx += 256) {
    int n = idx >> 6, k = idx & 63;
    w1s[n * WST + k] = (_Float16)((k < PE_DIM) ? w1[n * PE_DIM + k] : 0.0f);
    w2s[n * WST + k] = (_Float16)w2[idx];
  }
  for (int idx = t; idx < 32 * 64; idx += 256) {
    int n = idx >> 6, k = idx & 63;
    w3s[n * WST + k] = (_Float16)((n < DB) ? w3[n * HID + k] : 0.0f);
  }
  for (int idx = t; idx < 32 * 32; idx += 256) {
    int s = idx >> 5, d = idx & 31;
    float v = 0.0f;
    if (d < DB && s <= MC) {
      float m = mu[b * DB + d];
      v = (s < MC) ? fmaf(stdv[b * DB + d], eps[((size_t)s * BN + b) * DB + d], m)
                   : m;                 // s == 16 -> mean row uses mu_h
    }
    hss[s * HSST + d] = (_Float16)v;
  }
  if (t < 64) { b1s[t] = b1[t]; b2s[t] = b2[t]; }
  if (t < 32) b3s[t] = (t < DB) ? b3[t] : 0.0f;
  __syncthreads();

  int wv = t >> 5, lane = t & 31;
  int half = lane >> 4, mrow = lane & 15;
  int p0 = tile * 128 + wv * 16;        // base point for this wave
  int sd = shapes[b * 3 + 0], sh = shapes[b * 3 + 1], sw = shapes[b * 3 + 2];

  // ---- Fourier features directly in A-fragment layout ----
  int p = p0 + mrow;
  int ix = p & 31, iy = (p >> 5) & 31, iz = p >> 10;
  float c0 = axis_coord(iz, sd), c1 = axis_coord(iy, sh), c2 = axis_coord(ix, sw);
  v16h a0, a1;
#pragma unroll
  for (int i = 0; i < 16; ++i) {
    int k = half * 16 + i;   // half*16+i
    a0[i] = (_Float16)pe_val(c0, c1, c2, k);
    a1[i] = (_Float16)pe_val(c0, c1, c2, k + 32);
  }

  // ---- Layer 1: pe(16x64) @ w1^T -> h1(16x64), GELU ----
  {
    v8f acc[4] = {};
#pragma unroll
    for (int nt = 0; nt < 4; ++nt) {
      const _Float16* r = &w1s[(nt * 16 + mrow) * WST];
      acc[nt] = wmma16(a0, ld_contig16(r + half * 16), acc[nt]);
      acc[nt] = wmma16(a1, ld_contig16(r + 32 + half * 16), acc[nt]);
    }
#pragma unroll
    for (int nt = 0; nt < 4; ++nt) {
      int n = nt * 16 + mrow;
      float bias = b1s[n];
#pragma unroll
      for (int i = 0; i < 8; ++i)
        hstage[wv][(i + half * 8) * WST + n] = (_Float16)gelu_f(acc[nt][i] + bias);
    }
  }
  __syncthreads();

  // ---- Layer 2: h1 @ w2^T -> h2(16x64), GELU (hstage reused in place) ----
  {
    const _Float16* ar = &hstage[wv][mrow * WST];
    v16h A0 = ld_afrag(ar, half, 0);
    v16h A1 = ld_afrag(ar, half, 32);
    __syncthreads();   // all reads of h1 done before overwriting with h2
    v8f acc[4] = {};
#pragma unroll
    for (int nt = 0; nt < 4; ++nt) {
      const _Float16* r = &w2s[(nt * 16 + mrow) * WST];
      acc[nt] = wmma16(A0, ld_contig16(r + half * 16), acc[nt]);
      acc[nt] = wmma16(A1, ld_contig16(r + 32 + half * 16), acc[nt]);
    }
#pragma unroll
    for (int nt = 0; nt < 4; ++nt) {
      int n = nt * 16 + mrow;
      float bias = b2s[n];
#pragma unroll
      for (int i = 0; i < 8; ++i)
        hstage[wv][(i + half * 8) * WST + n] = (_Float16)gelu_f(acc[nt][i] + bias);
    }
  }
  __syncthreads();

  // ---- Layer 3: h2 @ w3^T + b3 -> Wflat(16x32, cols>=29 zero) ----
  {
    const _Float16* ar = &hstage[wv][mrow * WST];
    v16h A0 = ld_afrag(ar, half, 0);
    v16h A1 = ld_afrag(ar, half, 32);
    v8f d0 = {}, d1 = {};
    {
      const _Float16* r = &w3s[mrow * WST];
      d0 = wmma16(A0, ld_contig16(r + half * 16), d0);
      d0 = wmma16(A1, ld_contig16(r + 32 + half * 16), d0);
    }
    {
      const _Float16* r = &w3s[(16 + mrow) * WST];
      d1 = wmma16(A0, ld_contig16(r + half * 16), d1);
      d1 = wmma16(A1, ld_contig16(r + 32 + half * 16), d1);
    }
    float bb0 = b3s[mrow], bb1 = b3s[16 + mrow];
#pragma unroll
    for (int i = 0; i < 8; ++i) {
      int m = i + half * 8;
      wfs[wv][m * WFST + mrow]      = (_Float16)(d0[i] + bb0);
      wfs[wv][m * WFST + 16 + mrow] = (_Float16)(d1[i] + bb1);  // rows>=29 -> 0
    }
  }
  __syncthreads();

  // ---- Logits: Wflat(16x32) @ h_samples^T(32x17) via WMMA, mask + stage ----
  float* lls = (float*)(void*)&hstage[wv][0];   // 17*LST floats, buffer now free
  {
    const _Float16* ar = &wfs[wv][mrow * WFST];
    v16h AW = ld_afrag(ar, half, 0);
    v8f l0 = {}, l1 = {};
    l0 = wmma16(AW, ld_contig16(&hss[mrow * HSST] + half * 16), l0);
    l1 = wmma16(AW, ld_contig16(&hss[(16 + mrow) * HSST] + half * 16), l1);
    float lb = logb[0];
#pragma unroll
    for (int i = 0; i < 8; ++i) {
      int m = i + half * 8;
      int q = p0 + m;
      int qx = q & 31, qy = (q >> 5) & 31, qz = q >> 10;
      bool reg = (qz < sd) && (qy < sh) && (qx < sw);
      lls[mrow * LST + m] = reg ? (l0[i] + lb) : -INFINITY;   // s = mrow
      if (mrow == 0)                                          // s = 16 (mean)
        lls[16 * LST + m] = reg ? (l1[i] + lb) : -INFINITY;
    }
  }
  __syncthreads();

  // ---- coalesced global stores: 17 rows x 16 points, 64B per half-wave ----
  {
    int col = lane & 15, rsel = lane >> 4;
    size_t bOff = (size_t)b * KTOT + (size_t)p0;
#pragma unroll
    for (int rr = 0; rr < 18; rr += 2) {
      int r = rr + rsel;
      if (r < 17)
        out[(size_t)r * BKTOT + bOff + col] = lls[r * LST + col];
    }
  }
}

// ---------------------------------------------------------------------------
// Kernel C: in-place masked softmax over K=8192 per (s,b) row (17*B rows).
// Row cached in 32KB LDS: one read + one write of the 142MB tensor.
// ---------------------------------------------------------------------------
__global__ __launch_bounds__(256) void k_softmax(float* __restrict__ out) {
  __shared__ float row[KTOT];
  __shared__ float red[256];
  int t = threadIdx.x;
  size_t base = (size_t)blockIdx.x * KTOT;
  float mx = -INFINITY;
#pragma unroll
  for (int j = 0; j < KTOT / 256; ++j) {
    int i = t + j * 256;
    float v = out[base + i];
    row[i] = v;
    mx = fmaxf(mx, v);
  }
  red[t] = mx;
  __syncthreads();
  for (int off = 128; off > 0; off >>= 1) {
    if (t < off) red[t] = fmaxf(red[t], red[t + off]);
    __syncthreads();
  }
  mx = red[0];
  __syncthreads();
  float sum = 0.0f;
#pragma unroll
  for (int j = 0; j < KTOT / 256; ++j) {
    int i = t + j * 256;
    float e = expf(row[i] - mx);
    row[i] = e;
    sum += e;
  }
  red[t] = sum;
  __syncthreads();
  for (int off = 128; off > 0; off >>= 1) {
    if (t < off) red[t] += red[t + off];
    __syncthreads();
  }
  float inv = (red[0] > 0.0f) ? (1.0f / red[0]) : 0.0f;
#pragma unroll
  for (int j = 0; j < KTOT / 256; ++j) {
    int i = t + j * 256;
    out[base + i] = row[i] * inv;
  }
}

// ---------------------------------------------------------------------------
extern "C" void kernel_launch(void* const* d_in, const int* in_sizes, int n_in,
                              void* d_out, int out_size, void* d_ws, size_t ws_size,
                              hipStream_t stream) {
  (void)in_sizes; (void)n_in; (void)out_size; (void)d_ws; (void)ws_size;
  const float* features = (const float*)d_in[0];
  const float* eps      = (const float*)d_in[1];
  const float* mu_W     = (const float*)d_in[2];
  const float* mu_b     = (const float*)d_in[3];
  const float* std_W    = (const float*)d_in[4];
  const float* std_b    = (const float*)d_in[5];
  const float* w1       = (const float*)d_in[6];
  const float* b1       = (const float*)d_in[7];
  const float* w2       = (const float*)d_in[8];
  const float* b2       = (const float*)d_in[9];
  const float* w3       = (const float*)d_in[10];
  const float* b3       = (const float*)d_in[11];
  const float* logb     = (const float*)d_in[12];
  const int*   shapes   = (const int*)d_in[13];
  // d_in[14] (mask) unused: recomputed from shapes. d_in[15] (mc) fixed = 16.

  float* out = (float*)d_out;
  float* mu_out  = out + 17 * BKTOT;       // after zp + mean_profile
  float* std_out = mu_out + BN * DB;

  k_mustd<<<dim3(BN * 32), dim3(64), 0, stream>>>(features, mu_W, mu_b, std_W,
                                                  std_b, mu_out, std_out);
  k_mlp_logits<<<dim3(BN * (KTOT / 128)), dim3(256), 0, stream>>>(
      w1, b1, w2, b2, w3, b3, eps, shapes, logb, mu_out, std_out, out);
  k_softmax<<<dim3(17 * BN), dim3(256), 0, stream>>>(out);
}
